// LSTM_6081673691196
// MI455X (gfx1250) — matrix operand order repaired and checked by reading, not compiled, not run on hardware
//
#include <hip/hip_runtime.h>
#include <hip/hip_bf16.h>
#include <cmath>

// ---------------------------------------------------------------------------
// CDNA5 (gfx1250) LSTM: bf16 WMMA GEMMs, fp32 accumulate, persistent
// recurrence kernel with register-resident cell state and a grid barrier.
// ---------------------------------------------------------------------------

typedef __bf16 bf16_t;
typedef __attribute__((ext_vector_type(8)))  __bf16 v8bf;
typedef __attribute__((ext_vector_type(16))) __bf16 v16bf;
typedef __attribute__((ext_vector_type(8)))  float  v8f;

#define WMMA_BF16(a, b, c) \
  __builtin_amdgcn_wmma_f32_16x16x32_bf16(false, (a), false, (b), (short)0, (c), false, false)

// A fragment: 16x32 bf16 tile from row-major [M,K] storage.
// lane L (0..15): row L, K = k0+0..7 (v0-3) and k0+16..23 (v4-7)
// lane L+16     : row L, K = k0+8..15 and k0+24..31
__device__ __forceinline__ v16bf load_a_frag(const bf16_t* __restrict__ tile,
                                             int ld, int k0, int lane) {
  int row  = lane & 15;
  int koff = (lane >> 4) << 3;  // 0 or 8
  const bf16_t* p = tile + (size_t)row * ld + k0 + koff;
  v8bf lo = *(const v8bf*)(p);
  v8bf hi = *(const v8bf*)(p + 16);
  return __builtin_shufflevector(lo, hi, 0,1,2,3,4,5,6,7,8,9,10,11,12,13,14,15);
}

// B fragment: 32x16 bf16 tile; B stored transposed as row-major [N,K].
// lane L (0..15): column L, K = k0+0..7 and k0+16..23 ; lane L+16: column L, K = k0+8..15 and k0+24..31
__device__ __forceinline__ v16bf load_b_frag(const bf16_t* __restrict__ tileT,
                                             int ld, int k0, int lane) {
  int col  = lane & 15;
  int koff = (lane >> 4) << 3;  // 0 or 8
  const bf16_t* p = tileT + (size_t)col * ld + k0 + koff;
  v8bf lo = *(const v8bf*)(p);
  v8bf hi = *(const v8bf*)(p + 16);
  return __builtin_shufflevector(lo, hi, 0,1,2,3,4,5,6,7,8,9,10,11,12,13,14,15);
}

__device__ __forceinline__ float sigm(float x) { return 1.0f / (1.0f + __expf(-x)); }

// Sense-reversing grid barrier (32 co-resident blocks).
__device__ __forceinline__ void grid_sync(int* ctr, int* gen, int nBlocks) {
  __syncthreads();
  __threadfence();  // release h stores to device scope
  if (threadIdx.x == 0) {
    int g = __hip_atomic_load(gen, __ATOMIC_ACQUIRE, __HIP_MEMORY_SCOPE_AGENT);
    int old = __hip_atomic_fetch_add(ctr, 1, __ATOMIC_ACQ_REL, __HIP_MEMORY_SCOPE_AGENT);
    if (old == nBlocks - 1) {
      __hip_atomic_store(ctr, 0, __ATOMIC_RELAXED, __HIP_MEMORY_SCOPE_AGENT);
      __hip_atomic_fetch_add(gen, 1, __ATOMIC_ACQ_REL, __HIP_MEMORY_SCOPE_AGENT);
    } else {
      while (__hip_atomic_load(gen, __ATOMIC_ACQUIRE, __HIP_MEMORY_SCOPE_AGENT) == g)
        __builtin_amdgcn_s_sleep(2);
    }
  }
  __syncthreads();
  __threadfence();  // acquire: invalidate stale near caches before reading h
}

// ---------------------------------------------------------------------------
__global__ __launch_bounds__(256) void k_f32_to_bf16(const float* __restrict__ s,
                                                     bf16_t* __restrict__ d, size_t n) {
  size_t i = (size_t)blockIdx.x * blockDim.x + threadIdx.x;
  size_t stride = (size_t)gridDim.x * blockDim.x;
  for (; i < n; i += stride) d[i] = (bf16_t)s[i];
}

__global__ void k_zero_ints(int* __restrict__ p, int n) {
  int i = (int)threadIdx.x;
  if (i < n) p[i] = 0;
}

// ---------------------------------------------------------------------------
// Input projection: out[M,N](bf16) = A[M,K](bf16) @ Bt[N,K]^T (bf16) + bias[N].
// One wave per 32x64 output tile: 2 A frags x 4 B frags -> 8 WMMA : 12 loads.
__global__ __launch_bounds__(256) void k_in_proj(const bf16_t* __restrict__ A,
                                                 const bf16_t* __restrict__ Bt,
                                                 const float* __restrict__ bias,
                                                 bf16_t* __restrict__ out,
                                                 int M, int N, int K) {
  int lane = threadIdx.x & 31;
  int wave = (int)((blockIdx.x * blockDim.x + threadIdx.x) >> 5);
  int nBlkCnt = N >> 6;                      // 64-wide column blocks
  int mTile = wave / nBlkCnt;                // 32-row tiles
  int nBlk  = wave - mTile * nBlkCnt;
  if (mTile * 32 >= M) return;

  const bf16_t* A0 = A + (size_t)mTile * 32 * K;
  const bf16_t* A1 = A0 + (size_t)16 * K;
  const bf16_t* B0 = Bt + (size_t)(nBlk * 64) * K;
  const bf16_t* B1 = B0 + (size_t)16 * K;
  const bf16_t* B2 = B0 + (size_t)32 * K;
  const bf16_t* B3 = B0 + (size_t)48 * K;

  v8f a00 = {}, a01 = {}, a02 = {}, a03 = {};
  v8f a10 = {}, a11 = {}, a12 = {}, a13 = {};
  for (int k0 = 0; k0 < K; k0 += 32) {
    v16bf fa0 = load_a_frag(A0, K, k0, lane);
    v16bf fa1 = load_a_frag(A1, K, k0, lane);
    v16bf fb0 = load_b_frag(B0, K, k0, lane);
    v16bf fb1 = load_b_frag(B1, K, k0, lane);
    v16bf fb2 = load_b_frag(B2, K, k0, lane);
    v16bf fb3 = load_b_frag(B3, K, k0, lane);
    a00 = WMMA_BF16(fa0, fb0, a00);  a10 = WMMA_BF16(fa1, fb0, a10);
    a01 = WMMA_BF16(fa0, fb1, a01);  a11 = WMMA_BF16(fa1, fb1, a11);
    a02 = WMMA_BF16(fa0, fb2, a02);  a12 = WMMA_BF16(fa1, fb2, a12);
    a03 = WMMA_BF16(fa0, fb3, a03);  a13 = WMMA_BF16(fa1, fb3, a13);
  }

  int col = nBlk * 64 + (lane & 15);
  float b0 = bias[col], b1 = bias[col + 16], b2 = bias[col + 32], b3 = bias[col + 48];
  int r0 = mTile * 32 + ((lane >> 4) << 3);
  int r1 = r0 + 16;
#pragma unroll
  for (int r = 0; r < 8; ++r) {
    size_t o0 = (size_t)(r0 + r) * N + col;
    size_t o1 = (size_t)(r1 + r) * N + col;
    out[o0]      = (bf16_t)(a00[r] + b0);
    out[o0 + 16] = (bf16_t)(a01[r] + b1);
    out[o0 + 32] = (bf16_t)(a02[r] + b2);
    out[o0 + 48] = (bf16_t)(a03[r] + b3);
    out[o1]      = (bf16_t)(a10[r] + b0);
    out[o1 + 16] = (bf16_t)(a11[r] + b1);
    out[o1 + 32] = (bf16_t)(a12[r] + b2);
    out[o1 + 48] = (bf16_t)(a13[r] + b3);
  }
}

// ---------------------------------------------------------------------------
// Persistent per-layer recurrence: ONE launch runs all S timesteps.
// 256 waves (32 blocks x 8); each wave owns a fixed 16x16 h-tile and keeps its
// cell state in registers for the whole layer. Whh (8 MB bf16) is L2-resident.
// h0 == 0 is realized by skipping the GEMM at t==0.
__global__ __launch_bounds__(256) void k_recurrence(const bf16_t* __restrict__ Whh, // [4H,H]
                                                    const bf16_t* __restrict__ xp,  // [S,B,4H] bf16
                                                    bf16_t* __restrict__ y,         // [S,B,H]
                                                    float* __restrict__ hf32,       // [B,H]
                                                    int S_, int Bsz, int Hd,
                                                    int* __restrict__ ctr,
                                                    int* __restrict__ gen) {
  int lane = threadIdx.x & 31;
  int wave = (int)((blockIdx.x * blockDim.x + threadIdx.x) >> 5);
  int nTiles = Hd >> 4;                     // 64 column tiles
  int mTile = wave / nTiles;                // 0..3
  int nTile = wave - mTile * nTiles;

  const bf16_t* Bi = Whh + ((size_t)0 * Hd + nTile * 16) * Hd;
  const bf16_t* Bf = Whh + ((size_t)1 * Hd + nTile * 16) * Hd;
  const bf16_t* Bg = Whh + ((size_t)2 * Hd + nTile * 16) * Hd;
  const bf16_t* Bo = Whh + ((size_t)3 * Hd + nTile * 16) * Hd;

  int hcol  = nTile * 16 + (lane & 15);
  int rbase = mTile * 16 + ((lane >> 4) << 3);

  float creg[8];
#pragma unroll
  for (int r = 0; r < 8; ++r) creg[r] = 0.0f;

  for (int t = 0; t < S_; ++t) {
    v8f ai = {}, af = {}, ag = {}, ao = {};
    if (t > 0) {
      const bf16_t* Atile = y + ((size_t)(t - 1) * Bsz + mTile * 16) * Hd;
      for (int k0 = 0; k0 < Hd; k0 += 32) {
        v16bf a = load_a_frag(Atile, Hd, k0, lane);
        ai = WMMA_BF16(a, load_b_frag(Bi, Hd, k0, lane), ai);
        af = WMMA_BF16(a, load_b_frag(Bf, Hd, k0, lane), af);
        ag = WMMA_BF16(a, load_b_frag(Bg, Hd, k0, lane), ag);
        ao = WMMA_BF16(a, load_b_frag(Bo, Hd, k0, lane), ao);
      }
    }

    const bf16_t* xpt = xp + (size_t)t * Bsz * (4 * Hd);
    bf16_t* hout = y + (size_t)t * Bsz * Hd;
#pragma unroll
    for (int r = 0; r < 8; ++r) {
      int bi = rbase + r;
      const bf16_t* xr = xpt + (size_t)bi * (4 * Hd);
      float gi = ai[r] + (float)xr[hcol];
      float gf = af[r] + (float)xr[Hd + hcol];
      float gg = ag[r] + (float)xr[2 * Hd + hcol];
      float go = ao[r] + (float)xr[3 * Hd + hcol];
      float si = sigm(gi), sf = sigm(gf), so = sigm(go);
      float tg = tanhf(gg);
      float cn = sf * creg[r] + si * tg;     // cell state: registers only
      creg[r] = cn;
      float hn = so * tanhf(cn);
      hout[(size_t)bi * Hd + hcol] = (bf16_t)hn;
      if (t == S_ - 1) hf32[(size_t)bi * Hd + hcol] = hn;
    }

    grid_sync(ctr, gen, (int)gridDim.x);
  }
}

// ---------------------------------------------------------------------------
// FC head: out[B,NC] = h[B,H] @ fc_w[NC,H]^T + fc_b. Tiny (131 MFLOP), fp32.
__global__ __launch_bounds__(256) void k_fc(const float* __restrict__ h,
                                            const float* __restrict__ w,
                                            const float* __restrict__ b,
                                            float* __restrict__ out,
                                            int Bsz, int NCc, int Hd) {
  int idx = (int)(blockIdx.x * blockDim.x + threadIdx.x);
  if (idx >= Bsz * NCc) return;
  int bi = idx / NCc, n = idx - bi * NCc;
  const float* hr = h + (size_t)bi * Hd;
  const float* wr = w + (size_t)n * Hd;
  float s = b[n];
#pragma unroll 4
  for (int k = 0; k < Hd; ++k) s = fmaf(hr[k], wr[k], s);
  out[idx] = s;
}

// ---------------------------------------------------------------------------
extern "C" void kernel_launch(void* const* d_in, const int* in_sizes, int n_in,
                              void* d_out, int out_size, void* d_ws, size_t ws_size,
                              hipStream_t stream) {
  (void)in_sizes; (void)n_in; (void)out_size; (void)ws_size;
  constexpr int S = 512, Bz = 64, H = 1024, L = 4, NC = 1000;
  constexpr int H4 = 4 * H;

  const float* x    = (const float*)d_in[0];
  const float* W    = (const float*)d_in[1];
  const float* bia  = (const float*)d_in[2];
  const float* fc_w = (const float*)d_in[3];
  const float* fc_b = (const float*)d_in[4];
  float* out = (float*)d_out;

  char* p = (char*)d_ws;
  auto alloc = [&](size_t bytes) -> char* {
    char* q = p;
    p += (bytes + 255) & ~(size_t)255;
    return q;
  };
  bf16_t* x_bf = (bf16_t*)alloc((size_t)S * Bz * H * sizeof(bf16_t));       // 64 MB
  bf16_t* W_bf = (bf16_t*)alloc((size_t)L * 2 * H4 * H * sizeof(bf16_t));   // 64 MB
  bf16_t* xp   = (bf16_t*)alloc((size_t)S * Bz * H4 * sizeof(bf16_t));      // 256 MB
  bf16_t* yA   = (bf16_t*)alloc((size_t)S * Bz * H * sizeof(bf16_t));       // 64 MB
  bf16_t* yB   = (bf16_t*)alloc((size_t)S * Bz * H * sizeof(bf16_t));       // 64 MB
  float*  hf32 = (float*) alloc((size_t)Bz * H * sizeof(float));
  int*    bar  = (int*)   alloc(2 * sizeof(int));                            // {ctr, gen}

  // Barrier counter must start at 0 (ws is poisoned); gen can be anything.
  k_zero_ints<<<1, 32, 0, stream>>>(bar, 2);

  // One-time precision conversion of activations-in and weights.
  k_f32_to_bf16<<<2048, 256, 0, stream>>>(x, x_bf, (size_t)S * Bz * H);
  k_f32_to_bf16<<<2048, 256, 0, stream>>>(W, W_bf, (size_t)L * 2 * H4 * H);

  const bf16_t* in_seq = x_bf;
  bf16_t* out_seq = yA;
  for (int l = 0; l < L; ++l) {
    const bf16_t* Wih = W_bf + (size_t)l * 2 * H4 * H;
    const bf16_t* Whh = Wih + (size_t)H4 * H;
    const float*  bl  = bia + (size_t)l * H4;

    // Batched input projection: [S*B,4H] = [S*B,H] @ Wih^T + b  (bf16 out)
    {
      int M = S * Bz, N = H4, K = H;
      int waves  = (M / 32) * (N / 64);   // 65536 waves
      int blocks = waves * 32 / 256;      // 8192 blocks
      k_in_proj<<<blocks, 256, 0, stream>>>(in_seq, Wih, bl, xp, M, N, K);
    }

    // Entire recurrence for this layer in ONE persistent kernel launch.
    k_recurrence<<<32, 256, 0, stream>>>(Whh, xp, out_seq, hf32, S, Bz, H,
                                         bar + 0, bar + 1);

    in_seq = out_seq;
    out_seq = (out_seq == yA) ? yB : yA;
  }

  // Classifier head on the final hidden state (fp32).
  {
    int total = Bz * NC;
    k_fc<<<(total + 255) / 256, 256, 0, stream>>>(hf32, fc_w, fc_b, out, Bz, NC, H);
  }
}
